// Attention1D_14946486190324
// MI455X (gfx1250) — matrix operand order repaired and checked by reading, not compiled, not run on hardware
//
#include <hip/hip_runtime.h>
#include <hip/hip_bf16.h>

// ---------------------------------------------------------------------------
// Attention1D (XCiT-style cross-covariance attention) for MI455X / gfx1250
// bf16 WMMA pipeline, all GEMM operands K-contiguous (no strided gathers),
// 64x32 macro-tiles (8 f32x8 accumulators -> no VGPR spills).
// ---------------------------------------------------------------------------

typedef __attribute__((ext_vector_type(16))) __bf16 bf16x16;
typedef __attribute__((ext_vector_type(8)))  __bf16 bf16x8;
typedef __attribute__((ext_vector_type(8)))  float  f32x8;

#define DEVI __device__ __forceinline__

constexpr int B  = 8;
constexpr int C  = 1024;
constexpr int T  = 4096;
constexpr int H  = 16;
constexpr int DH = 64;          // head dim
constexpr int C3 = 3 * C;       // 3072

// ---------------------------------------------------------------------------
// WMMA 64x32 macro-tile: D(64x32) += A(64xK) * B(Kx32), bf16 in, f32 accum.
//   A: row-major [M][K], K contiguous, lda elements.
//   B: element (k,n) at Bm[n*ldb + k]  (K contiguous per column).
// A-frag (16-bit 16x32): lane<16 -> row M=lane, K {0..7,16..23};
//                        lane>=16 -> row M=lane&15, K {8..15,24..31}.
// B-frag: lane holds col N=lane&15; lanes<16 K {0..7,16..23}, lanes>=16 K {8..15,24..31}.
// ---------------------------------------------------------------------------
DEVI void mma64x32(const __bf16* __restrict__ A, int lda,
                   const __bf16* __restrict__ Bm, int ldb,
                   int row0, int col0, int K, f32x8 (&acc)[4][2]) {
  const int lane = threadIdx.x & 31;
  const int half = lane >> 4;
  const int l15  = lane & 15;
  for (int k0 = 0; k0 < K; k0 += 32) {
    if (k0 + 32 < K) {
      __builtin_prefetch(A + (size_t)(row0 + l15) * lda + k0 + 32, 0, 1);
      __builtin_prefetch(Bm + (size_t)(col0 + l15) * ldb + k0 + 32, 0, 1);
    }
    bf16x16 af[4];
#pragma unroll
    for (int i = 0; i < 4; ++i) {
      const __bf16* p = A + (size_t)(row0 + 16 * i + l15) * lda + k0 + half * 8;
      bf16x8 lo = *(const bf16x8*)(p);
      bf16x8 hi = *(const bf16x8*)(p + 16);
      bf16x16 v;
#pragma unroll
      for (int e = 0; e < 8; ++e) { v[e] = lo[e]; v[e + 8] = hi[e]; }
      af[i] = v;
    }
    bf16x16 bfb[2];
#pragma unroll
    for (int j = 0; j < 2; ++j) {
      const __bf16* q = Bm + (size_t)(col0 + 16 * j + l15) * ldb + k0 + half * 8;
      bf16x8 lo = *(const bf16x8*)(q);
      bf16x8 hi = *(const bf16x8*)(q + 16);
      bf16x16 v;
#pragma unroll
      for (int e = 0; e < 8; ++e) { v[e] = lo[e]; v[e + 8] = hi[e]; }
      bfb[j] = v;
    }
#pragma unroll
    for (int i = 0; i < 4; ++i)
#pragma unroll
      for (int j = 0; j < 2; ++j)
        acc[i][j] = __builtin_amdgcn_wmma_f32_16x16x32_bf16(
            false, af[i], false, bfb[j], (short)0, acc[i][j], false, false);
  }
}

DEVI void zero_acc(f32x8 (&acc)[4][2]) {
#pragma unroll
  for (int i = 0; i < 4; ++i)
#pragma unroll
    for (int j = 0; j < 2; ++j) {
      f32x8 z = {};
      acc[i][j] = z;
    }
}

// C/D 16x16 f32 layout: VGPR e -> M = e + (lane>=16 ? 8 : 0), N = lane&15.
DEVI void store_f32(float* __restrict__ Dst, int ldd, int row0, int col0,
                    const f32x8 (&acc)[4][2], const float* __restrict__ bias) {
  const int lane = threadIdx.x & 31;
  const int half = lane >> 4, l15 = lane & 15;
#pragma unroll
  for (int i = 0; i < 4; ++i)
#pragma unroll
    for (int j = 0; j < 2; ++j)
#pragma unroll
      for (int e = 0; e < 8; ++e) {
        int r = row0 + 16 * i + half * 8 + e;
        int c = col0 + 16 * j + l15;
        float v = acc[i][j][e];
        if (bias) v += bias[r];
        Dst[(size_t)r * ldd + c] = v;
      }
}

DEVI void store_f32_atomic(float* __restrict__ Dst, int ldd, int row0, int col0,
                           const f32x8 (&acc)[4][2]) {
  const int lane = threadIdx.x & 31;
  const int half = lane >> 4, l15 = lane & 15;
#pragma unroll
  for (int i = 0; i < 4; ++i)
#pragma unroll
    for (int j = 0; j < 2; ++j)
#pragma unroll
      for (int e = 0; e < 8; ++e) {
        int r = row0 + 16 * i + half * 8 + e;
        int c = col0 + 16 * j + l15;
        atomicAdd(&Dst[(size_t)r * ldd + c], acc[i][j][e]);
      }
}

// Transposed store: Dst[col][row] (row dim contiguous) -> one bf16x8 per frag.
DEVI void store_bf16_T(__bf16* __restrict__ Dst, int ldd, int row0, int col0,
                       const f32x8 (&acc)[4][2]) {
  const int lane = threadIdx.x & 31;
  const int half = lane >> 4, l15 = lane & 15;
#pragma unroll
  for (int i = 0; i < 4; ++i)
#pragma unroll
    for (int j = 0; j < 2; ++j) {
      int r = row0 + 16 * i + half * 8;
      int c = col0 + 16 * j + l15;
      bf16x8 v;
#pragma unroll
      for (int e = 0; e < 8; ++e) v[e] = (__bf16)acc[i][j][e];
      *(bf16x8*)(Dst + (size_t)c * ldd + r) = v;
    }
}

// ---------------------------------------------------------------------------
// Elementwise / transpose kernels
// ---------------------------------------------------------------------------

__global__ void k_cvt_bf16(const float* __restrict__ s, __bf16* __restrict__ d,
                           size_t n) {
  size_t i = (size_t)blockIdx.x * blockDim.x + threadIdx.x;
  size_t stride = (size_t)gridDim.x * blockDim.x;
  for (; i < n; i += stride) d[i] = (__bf16)s[i];
}

__global__ void k_zero_f32(float* __restrict__ p, size_t n) {
  size_t i = (size_t)blockIdx.x * blockDim.x + threadIdx.x;
  size_t stride = (size_t)gridDim.x * blockDim.x;
  for (; i < n; i += stride) p[i] = 0.0f;
}

// Tiled 32x32 transpose (+ convert to bf16): S[bat][r][c] -> D[bat][c][r]
template<typename TS>
__global__ void k_txp_to_bf16(const TS* __restrict__ S, __bf16* __restrict__ D,
                              int rows, int cols, size_t sStride,
                              size_t dStride) {
  __shared__ __bf16 tile[32][33];
  int tilesPerBatch = (rows / 32) * (cols / 32);
  int bat = blockIdx.x / tilesPerBatch;
  int rem = blockIdx.x % tilesPerBatch;
  int rT = rem / (cols / 32), cT = rem % (cols / 32);
  int r0 = rT * 32, c0 = cT * 32;
  const TS* Sb = S + (size_t)bat * sStride;
  __bf16* Db = D + (size_t)bat * dStride;
  int tx = threadIdx.x & 31, ty = threadIdx.x >> 5;      // 32 x 8
#pragma unroll
  for (int rr = 0; rr < 32; rr += 8)
    tile[ty + rr][tx] = (__bf16)Sb[(size_t)(r0 + ty + rr) * cols + c0 + tx];
  __syncthreads();
#pragma unroll
  for (int rr = 0; rr < 32; rr += 8)
    Db[(size_t)(c0 + ty + rr) * rows + r0 + tx] = tile[tx][ty + rr];
}

// ---------------------------------------------------------------------------
// GEMM kernels
// ---------------------------------------------------------------------------

// qkv[b] (3072 x 4096) = Wq(3072x1024) * x_b(1024x4096) + b_qkv
//   B operand: xT[b][t][c] -> (k=c, n=t) at xT[n*C + k]  (K contiguous)
__global__ void k_gemm_qkv(const __bf16* __restrict__ Wq,
                           const __bf16* __restrict__ XT,
                           const float* __restrict__ bias,
                           float* __restrict__ Q) {
  const int MT = C3 / 64, NT = T / 32;   // 48, 128
  int w = blockIdx.x * (blockDim.x >> 5) + (threadIdx.x >> 5);
  if (w >= B * MT * NT) return;
  int b = w / (MT * NT);
  int rem = w % (MT * NT);
  int mt = rem / NT, nt = rem % NT;
  f32x8 acc[4][2];
  zero_acc(acc);
  mma64x32(Wq, C, XT + (size_t)b * T * C, C, mt * 64, nt * 32, C, acc);
  store_f32(Q + (size_t)b * C3 * T, T, mt * 64, nt * 32, acc, bias);
}

// fused: depthwise conv(k=3,pad=1)+bias, L2-normalize q/k over T, emit bf16
__global__ void k_dwnorm(const float* __restrict__ qkv,
                         const float* __restrict__ wdw,
                         const float* __restrict__ bdw,
                         __bf16* __restrict__ qn, __bf16* __restrict__ kn,
                         __bf16* __restrict__ vb) {
  __shared__ float red[256];
  int row = blockIdx.x;               // [0, B*C3)
  int b = row / C3, ch = row % C3;
  const float* src = qkv + (size_t)row * T;
  float w0 = wdw[ch * 3 + 0], w1 = wdw[ch * 3 + 1], w2 = wdw[ch * 3 + 2];
  float bb = bdw[ch];
  bool isqk = ch < 2 * C;
  float vals[16];
  float ss = 0.f;
  int t0 = threadIdx.x * 16;
#pragma unroll
  for (int j = 0; j < 16; ++j) {
    int t = t0 + j;
    float xm = (t > 0) ? src[t - 1] : 0.f;
    float xc = src[t];
    float xp = (t < T - 1) ? src[t + 1] : 0.f;
    float v = fmaf(w0, xm, fmaf(w1, xc, fmaf(w2, xp, bb)));
    vals[j] = v;
    ss += v * v;
  }
  red[threadIdx.x] = ss;
  __syncthreads();
  for (int s = 128; s > 0; s >>= 1) {
    if ((int)threadIdx.x < s) red[threadIdx.x] += red[threadIdx.x + s];
    __syncthreads();
  }
  float scale = 1.f;
  if (isqk) scale = 1.f / fmaxf(sqrtf(red[0]), 1e-12f);
  __bf16* dst;
  int c;
  if (ch < C)          { dst = qn; c = ch; }
  else if (ch < 2 * C) { dst = kn; c = ch - C; }
  else                 { dst = vb; c = ch - 2 * C; }
  __bf16* drow = dst + ((size_t)b * C + c) * T;
#pragma unroll
  for (int j = 0; j < 16; ++j) drow[t0 + j] = (__bf16)(vals[j] * scale);
}

// attn[bh] (64x64) += q[bh](64 x Tc) * k[bh]^T(Tc x 64); K split in 16 chunks,
// 2 N-tiles of 32. B operand: kn[e][t] -> (k=t, n=e) at kn[n*T + k] (contig).
__global__ void k_gemm_attn(const __bf16* __restrict__ qn,
                            const __bf16* __restrict__ kn,
                            float* __restrict__ attn) {
  const int KC = 16, KLEN = T / KC;    // 16 chunks of 256
  int w = blockIdx.x * (blockDim.x >> 5) + (threadIdx.x >> 5);
  if (w >= B * H * KC * 2) return;
  int nt = w & 1;
  int kc = (w >> 1) % KC;
  int bh = w / (2 * KC);
  const __bf16* A  = qn + (size_t)bh * DH * T + kc * KLEN;
  const __bf16* Bm = kn + (size_t)bh * DH * T + kc * KLEN;
  f32x8 acc[4][2];
  zero_acc(acc);
  mma64x32(A, T, Bm, T, 0, nt * 32, KLEN, acc);
  store_f32_atomic(attn + (size_t)bh * DH * DH, DH, 0, nt * 32, acc);
}

// softmax over last dim (64) with per-head temperature; one wave per row
__global__ void k_softmax(const float* __restrict__ attn,
                          const float* __restrict__ temp,
                          __bf16* __restrict__ P) {
  int w = blockIdx.x * (blockDim.x >> 5) + (threadIdx.x >> 5);
  if (w >= B * H * DH) return;
  int h = (w / DH) % H;
  int lane = threadIdx.x & 31;
  const float* r = attn + (size_t)w * DH;
  float tt = temp[h];
  float v0 = r[lane] * tt;
  float v1 = r[lane + 32] * tt;
  float m = fmaxf(v0, v1);
#pragma unroll
  for (int s = 16; s > 0; s >>= 1) m = fmaxf(m, __shfl_xor(m, s, 32));
  float e0 = __expf(v0 - m), e1 = __expf(v1 - m);
  float sum = e0 + e1;
#pragma unroll
  for (int s = 16; s > 0; s >>= 1) sum += __shfl_xor(sum, s, 32);
  float inv = 1.f / sum;
  __bf16* pr = P + (size_t)w * DH;
  pr[lane]      = (__bf16)(e0 * inv);
  pr[lane + 32] = (__bf16)(e1 * inv);
}

// O[bh] (64 x 4096) = P[bh](64x64) * v[bh](64x4096)
//   B operand: vT[bh][t][e] -> (k=e, n=t) at vT[n*64 + k]  (K contiguous)
//   Output stored transposed: outiT[b][t][h*64 + c_local]  (bf16x8 stores)
__global__ void k_gemm_out(const __bf16* __restrict__ P,
                           const __bf16* __restrict__ vT,
                           __bf16* __restrict__ outiT) {
  const int NT = T / 32;               // 128
  int w = blockIdx.x * (blockDim.x >> 5) + (threadIdx.x >> 5);
  if (w >= B * H * NT) return;
  int bh = w / NT, nt = w % NT;
  int b = bh / H, h = bh % H;
  f32x8 acc[4][2];
  zero_acc(acc);
  mma64x32(P + (size_t)bh * DH * DH, DH, vT + (size_t)bh * T * DH, DH,
           0, nt * 32, DH, acc);
  store_bf16_T(outiT + (size_t)b * T * C + h * DH, C, 0, nt * 32, acc);
}

// y[b] (1024x4096) = Wo(1024x1024) * out_b(1024x4096) + b_out
//   B operand: outiT[b][t][c] -> (k=c, n=t) at outiT[n*C + k]  (K contiguous)
__global__ void k_gemm_final(const __bf16* __restrict__ Wo,
                             const __bf16* __restrict__ OiT,
                             const float* __restrict__ bias,
                             float* __restrict__ Y) {
  const int MT = C / 64, NT = T / 32;  // 16, 128
  int w = blockIdx.x * (blockDim.x >> 5) + (threadIdx.x >> 5);
  if (w >= B * MT * NT) return;
  int b = w / (MT * NT);
  int rem = w % (MT * NT);
  int mt = rem / NT, nt = rem % NT;
  f32x8 acc[4][2];
  zero_acc(acc);
  mma64x32(Wo, C, OiT + (size_t)b * T * C, C, mt * 64, nt * 32, C, acc);
  store_f32(Y + (size_t)b * C * T, T, mt * 64, nt * 32, acc, bias);
}

// ---------------------------------------------------------------------------
// Launch
// ---------------------------------------------------------------------------
extern "C" void kernel_launch(void* const* d_in, const int* in_sizes, int n_in,
                              void* d_out, int out_size, void* d_ws,
                              size_t ws_size, hipStream_t stream) {
  const float* x     = (const float*)d_in[0];
  const float* w_qkv = (const float*)d_in[1];
  const float* b_qkv = (const float*)d_in[2];
  const float* w_dw  = (const float*)d_in[3];
  const float* b_dw  = (const float*)d_in[4];
  const float* w_out = (const float*)d_in[5];
  const float* b_out = (const float*)d_in[6];
  const float* temp  = (const float*)d_in[7];
  float* y = (float*)d_out;

  char* ws = (char*)d_ws;
  size_t off = 0;
  auto carve = [&](size_t bytes) {
    void* p = ws + off;
    off = (off + bytes + 255) & ~(size_t)255;
    return p;
  };
  __bf16* xT      = (__bf16*)carve((size_t)B * T * C * 2);       // 64 MB
  __bf16* wqkv_bf = (__bf16*)carve((size_t)C3 * C * 2);          // 6 MB
  __bf16* wout_bf = (__bf16*)carve((size_t)C * C * 2);           // 2 MB
  float*  qkv32   = (float*) carve((size_t)B * C3 * T * 4);      // 384 MB
  __bf16* qn      = (__bf16*)carve((size_t)B * C * T * 2);       // 64 MB
  __bf16* kn      = (__bf16*)carve((size_t)B * C * T * 2);       // 64 MB
  __bf16* vb      = (__bf16*)carve((size_t)B * C * T * 2);       // 64 MB
  __bf16* vT      = (__bf16*)carve((size_t)B * H * T * DH * 2);  // 64 MB
  float*  attn32  = (float*) carve((size_t)B * H * DH * DH * 4); // 2 MB
  __bf16* Pbf     = (__bf16*)carve((size_t)B * H * DH * DH * 2); // 1 MB
  __bf16* outiT   = (__bf16*)carve((size_t)B * T * C * 2);       // 64 MB
  (void)ws_size; (void)n_in; (void)in_sizes; (void)out_size;

  const int THR = 256;           // 8 waves of 32

  // x -> xT (bf16, transposed); weights -> bf16
  k_txp_to_bf16<float><<<B * (C / 32) * (T / 32), THR, 0, stream>>>(
      x, xT, C, T, (size_t)C * T, (size_t)T * C);
  k_cvt_bf16<<<1024, THR, 0, stream>>>(w_qkv, wqkv_bf, (size_t)C3 * C);
  k_cvt_bf16<<<512,  THR, 0, stream>>>(w_out, wout_bf, (size_t)C * C);

  // qkv GEMM: B * 48 * 128 = 49152 waves
  k_gemm_qkv<<<49152 / 8, THR, 0, stream>>>(wqkv_bf, xT, b_qkv, qkv32);

  // fused dwconv + l2norm + bf16 split
  k_dwnorm<<<B * C3, THR, 0, stream>>>(qkv32, w_dw, b_dw, qn, kn, vb);

  // v -> vT per (b,h): 64 x T -> T x 64
  k_txp_to_bf16<__bf16><<<B * H * (DH / 32) * (T / 32), THR, 0, stream>>>(
      vb, vT, DH, T, (size_t)DH * T, (size_t)T * DH);

  // attention scores (K-split atomic accumulation): 8*16*16*2 = 4096 waves
  k_zero_f32<<<2048, THR, 0, stream>>>(attn32, (size_t)B * H * DH * DH);
  k_gemm_attn<<<4096 / 8, THR, 0, stream>>>(qn, kn, attn32);

  // softmax (temp scale inside): 8192 waves
  k_softmax<<<8192 / 8, THR, 0, stream>>>(attn32, temp, Pbf);

  // attention output GEMM: 8*16*128 = 16384 waves (stores transposed)
  k_gemm_out<<<16384 / 8, THR, 0, stream>>>(Pbf, vT, outiT);

  // final projection: 8*16*128 = 16384 waves
  k_gemm_final<<<16384 / 8, THR, 0, stream>>>(wout_bf, outiT, b_out, y);
}
